// ResidualQuantizer_82643760710087
// MI455X (gfx1250) — matrix operand order repaired and checked by reading, not compiled, not run on hardware
//
#include <hip/hip_runtime.h>
#include <hip/hip_bf16.h>

typedef __attribute__((ext_vector_type(16))) __bf16        v16bf;
typedef __attribute__((ext_vector_type(8)))  float         v8f;
typedef __attribute__((ext_vector_type(4)))  float         f32x4;
typedef __attribute__((ext_vector_type(2)))  float         f32x2;
typedef __attribute__((ext_vector_type(4)))  unsigned int  u32x4;

#define N_ROWS        65536
#define DIM           128
#define K_CODES       1024
#define NCB           8
#define ROWS_PER_BLK  128
#define THREADS       256
#define STAGE_CODES   64
#define N_STAGES      (K_CODES / STAGE_CODES)   // 16
#define N_SUB         (STAGE_CODES / 16)        // 4
#define LDS_STRIDE    68                        // 64 data words + 4 pad (conflict-free B reads)
#define BUF_WORDS     (STAGE_CODES * LDS_STRIDE)
#define BUF_BYTES     (BUF_WORDS * 4)           // 17408

// output layout: z_q [65536*128] | codes [65536*8] | commit | codebook
#define OFF_CODES     (N_ROWS * DIM)                 // 8388608
#define OFF_COMMIT    (OFF_CODES + N_ROWS * NCB)     // 8912896
#define OFF_CBLOSS    (OFF_COMMIT + 1)               // 8912897

// workspace layout: packed-bf16 codebook [8192 codes * 64 u32] then norms [8192 f32]
#define WS_NORM_OFF_U32  (NCB * K_CODES * (DIM / 2))  // 524288 u32 = 2 MB

__device__ __forceinline__ unsigned int f2bf(float f) {
    unsigned int x = __float_as_uint(f);
    x += 0x7FFFu + ((x >> 16) & 1u);     // round-to-nearest-even
    return x >> 16;
}

__global__ void __launch_bounds__(THREADS) rq_init_kernel(float* out) {
    out[OFF_COMMIT] = 0.0f;
    out[OFF_CBLOSS] = 0.0f;
}

// one-shot: pack codebooks f32 -> bf16x2 (2 dims per u32)
__global__ void __launch_bounds__(THREADS) rq_prep_pack(const float* __restrict__ cb,
                                                        unsigned int* __restrict__ wsBF) {
    int idx = blockIdx.x * THREADS + threadIdx.x;       // 0 .. 524287
    const f32x2* s2 = (const f32x2*)cb;
    f32x2 xy = s2[idx];
    wsBF[idx] = f2bf(xy.x) | (f2bf(xy.y) << 16);
}

// one-shot: ||c||^2 per code (f32, matches reference precision)
__global__ void __launch_bounds__(THREADS) rq_prep_norm(const float* __restrict__ cb,
                                                        float* __restrict__ wsNorm) {
    int code = blockIdx.x * THREADS + threadIdx.x;      // 0 .. 8191
    const f32x4* r = (const f32x4*)(cb + (size_t)code * DIM);
    float s = 0.0f;
    #pragma unroll
    for (int k = 0; k < DIM / 4; ++k) {
        f32x4 v = r[k];
        s += v.x * v.x + v.y * v.y + v.z * v.z + v.w * v.w;
    }
    wsNorm[code] = s;
}

__global__ void __launch_bounds__(THREADS) rq_main_kernel(
        const float* __restrict__ z,            // [65536,128]
        const float* __restrict__ cb,           // [8,1024,128] f32 (for exact gather)
        const unsigned int* __restrict__ wsBF,  // packed bf16 codebooks
        const float* __restrict__ wsNorm,       // per-code norms
        float* __restrict__ out) {

    __shared__ unsigned int ldsCB[2][BUF_WORDS];   // double-buffered bf16 tile (34816 B)

    const int tid    = threadIdx.x;
    const int lane   = tid & 31;
    const int wave   = tid >> 5;          // 0..7
    const int laneLo = lane & 15;
    const int laneHi = lane >> 4;         // 0 or 1

    const int rowBase = blockIdx.x * ROWS_PER_BLK + wave * 16;
    const int myRow   = rowBase + laneLo; // lane L and L+16 both own row L (dims split)

    // ---- per-thread async-copy slice: 4 x b128 chunks per stage ----
    unsigned int ldsOff[4], gOff[4];
    #pragma unroll
    for (int k = 0; k < 4; ++k) {
        int q   = tid + k * THREADS;         // 0..1023
        int row = q >> 4;
        int col = q & 15;
        ldsOff[k] = (unsigned int)(row * (LDS_STRIDE * 4) + col * 16);
        gOff[k]   = (unsigned int)(row * 256 + col * 16);
    }
    const unsigned int ldsBase0 = (unsigned int)(unsigned long long)(void*)&ldsCB[0][0];

    auto issue_stage = [&](int buf, int cbIdx, int stage) {
        const unsigned int* gbase =
            wsBF + ((size_t)(cbIdx * K_CODES + stage * STAGE_CODES)) * (DIM / 2);
        unsigned int lbase = ldsBase0 + (unsigned int)(buf * BUF_BYTES);
        #pragma unroll
        for (int k = 0; k < 4; ++k) {
            unsigned int la = lbase + ldsOff[k];
            unsigned int ga = gOff[k];
            asm volatile("global_load_async_to_lds_b128 %0, %1, %2 offset:0"
                         :: "v"(la), "v"(ga), "s"(gbase)
                         : "memory");
        }
    };

    // ---- load residual = z in A-fragment dim ordering ----
    // resid[j], j = 16*c + 8*hh + e  <->  dim = 32*c + 16*hh + 8*laneHi + e
    float resid[64];
    {
        const f32x4* zr4 = (const f32x4*)(z + (size_t)myRow * DIM);
        #pragma unroll
        for (int c = 0; c < 4; ++c) {
            #pragma unroll
            for (int hh = 0; hh < 2; ++hh) {
                int f4 = 8 * c + 4 * hh + laneHi * 2;
                f32x4 a = zr4[f4];
                f32x4 b = zr4[f4 + 1];
                int j = 16 * c + 8 * hh;
                resid[j+0]=a.x; resid[j+1]=a.y; resid[j+2]=a.z; resid[j+3]=a.w;
                resid[j+4]=b.x; resid[j+5]=b.y; resid[j+6]=b.z; resid[j+7]=b.w;
            }
        }
    }

    float lossAcc = 0.0f;

    issue_stage(0, 0, 0);   // preload first tile

    #pragma unroll 1
    for (int i = 0; i < NCB; ++i) {
        // ---- bf16 A fragments from current residual ----
        v16bf afrag[4];
        #pragma unroll
        for (int c = 0; c < 4; ++c) {
            union { v16bf v; unsigned short u[16]; } av;
            #pragma unroll
            for (int j = 0; j < 16; ++j) av.u[j] = (unsigned short)f2bf(resid[16 * c + j]);
            afrag[c] = av.v;
        }

        float minv[8];
        int   mini[8];
        #pragma unroll
        for (int r = 0; r < 8; ++r) { minv[r] = 3.4e38f; mini[r] = 0x7fffffff; }

        const float* cbi    = cb + (size_t)i * K_CODES * DIM;
        const float* normsI = wsNorm + i * K_CODES;

        #pragma unroll 1
        for (int stage = 0; stage < N_STAGES; ++stage) {
            const int buf = stage & 1;
            // own async copies into buf complete; barrier makes every wave's copies visible
            asm volatile("s_wait_asynccnt 0x0" ::: "memory");
            __syncthreads();

            // kick next tile into the other buffer (overlaps with compute below)
            if (stage + 1 < N_STAGES)      issue_stage(buf ^ 1, i, stage + 1);
            else if (i + 1 < NCB)          issue_stage(buf ^ 1, i + 1, 0);

            const unsigned int* tile = buf ? &ldsCB[1][0] : &ldsCB[0][0];

            // ---- hand-pipelined subtiles: ping-pong B buffers keep 8 ds_load_b128
            //      in flight under each WMMA chain ----
            v16bf b0[4], b1[4];

            auto load_sub = [&](v16bf (&bf)[4], int sub) {
                #pragma unroll
                for (int kc = 0; kc < 4; ++kc) {
                    int wordBase = (sub * 16 + laneLo) * LDS_STRIDE + kc * 16 + laneHi * 4;
                    union { v16bf v; u32x4 q[2]; } b;
                    const u32x4* bp = (const u32x4*)&tile[wordBase];
                    b.q[0] = bp[0];
                    b.q[1] = bp[2];
                    bf[kc] = b.v;
                }
            };
            auto compute_sub = [&](const v16bf (&bf)[4], int sub) {
                v8f acc = {};
                #pragma unroll
                for (int kc = 0; kc < 4; ++kc)
                    acc = __builtin_amdgcn_wmma_f32_16x16x32_bf16(
                              false, afrag[kc], false, bf[kc], (short)0, acc, false, false);
                int   codeIdx = stage * STAGE_CODES + sub * 16 + laneLo;
                float cn      = normsI[codeIdx];          // L2-resident broadcast
                #pragma unroll
                for (int r = 0; r < 8; ++r) {
                    float val = fmaf(-2.0f, acc[r], cn);  // ||c||^2 - 2 r.c
                    bool take = (val < minv[r]) | ((val == minv[r]) & (codeIdx < mini[r]));
                    minv[r] = take ? val : minv[r];
                    mini[r] = take ? codeIdx : mini[r];
                }
            };

            load_sub(b0, 0);
            load_sub(b1, 1);
            compute_sub(b0, 0);
            load_sub(b0, 2);
            compute_sub(b1, 1);
            load_sub(b1, 3);
            compute_sub(b0, 2);
            compute_sub(b1, 3);
        }

        // ---- reduce argmin across the 16 columns in each lane-half ----
        #pragma unroll
        for (int r = 0; r < 8; ++r) {
            #pragma unroll
            for (int m = 1; m <= 8; m <<= 1) {
                float ov = __shfl_xor(minv[r], m, 32);
                int   oi = __shfl_xor(mini[r], m, 32);
                bool take = (ov < minv[r]) | ((ov == minv[r]) & (oi < mini[r]));
                minv[r] = take ? ov : minv[r];
                mini[r] = take ? oi : mini[r];
            }
        }

        // ---- broadcast each row's winner; lane owning row laneLo keeps it ----
        int myIdx = 0;
        #pragma unroll
        for (int r = 0; r < 16; ++r) {
            int bv = __shfl(mini[r & 7], (r < 8) ? 0 : 16, 32);
            if (laneLo == r) myIdx = bv;
        }

        if (lane < 16)
            out[OFF_CODES + (size_t)myRow * NCB + i] = (float)myIdx;

        // ---- gather winning code (f32, L2-resident), update residual + loss ----
        {
            const f32x4* cr4 = (const f32x4*)(cbi + (size_t)myIdx * DIM);
            #pragma unroll
            for (int c = 0; c < 4; ++c) {
                #pragma unroll
                for (int hh = 0; hh < 2; ++hh) {
                    int f4 = 8 * c + 4 * hh + laneHi * 2;
                    f32x4 a = cr4[f4];
                    f32x4 b = cr4[f4 + 1];
                    int j = 16 * c + 8 * hh;
                    resid[j+0]-=a.x; resid[j+1]-=a.y; resid[j+2]-=a.z; resid[j+3]-=a.w;
                    resid[j+4]-=b.x; resid[j+5]-=b.y; resid[j+6]-=b.z; resid[j+7]-=b.w;
                }
            }
            #pragma unroll
            for (int j = 0; j < 64; ++j) lossAcc += resid[j] * resid[j];
        }
    }

    // ---- z_q = z - final residual (straight-through output) ----
    {
        const f32x4* zr4 = (const f32x4*)(z + (size_t)myRow * DIM);
        f32x4*       or4 = (f32x4*)(out + (size_t)myRow * DIM);
        #pragma unroll
        for (int c = 0; c < 4; ++c) {
            #pragma unroll
            for (int hh = 0; hh < 2; ++hh) {
                int f4 = 8 * c + 4 * hh + laneHi * 2;
                f32x4 a = zr4[f4];
                f32x4 b = zr4[f4 + 1];
                int j = 16 * c + 8 * hh;
                f32x4 oa, ob;
                oa.x = a.x - resid[j+0]; oa.y = a.y - resid[j+1];
                oa.z = a.z - resid[j+2]; oa.w = a.w - resid[j+3];
                ob.x = b.x - resid[j+4]; ob.y = b.y - resid[j+5];
                ob.z = b.z - resid[j+6]; ob.w = b.w - resid[j+7];
                or4[f4]     = oa;
                or4[f4 + 1] = ob;
            }
        }
    }

    // ---- losses: commit == codebook == sum(resid_i^2)/(N*D) accumulated per iter ----
    #pragma unroll
    for (int m = 1; m <= 16; m <<= 1) lossAcc += __shfl_xor(lossAcc, m, 32);
    if (lane == 0) {
        const float inv = 1.0f / ((float)N_ROWS * (float)DIM);
        float v = lossAcc * inv;
        atomicAdd(out + OFF_COMMIT, v);
        atomicAdd(out + OFF_CBLOSS, v);
    }
}

extern "C" void kernel_launch(void* const* d_in, const int* in_sizes, int n_in,
                              void* d_out, int out_size, void* d_ws, size_t ws_size,
                              hipStream_t stream) {
    const float* z  = (const float*)d_in[0];   // [65536,128]
    const float* cb = (const float*)d_in[1];   // [8,1024,128]
    float* out      = (float*)d_out;

    unsigned int* wsBF = (unsigned int*)d_ws;                 // 2 MB packed bf16
    float*        wsNm = (float*)(wsBF + WS_NORM_OFF_U32);    // 32 KB norms

    rq_init_kernel<<<1, THREADS, 0, stream>>>(out);
    rq_prep_pack<<<(NCB * K_CODES * DIM / 2) / THREADS, THREADS, 0, stream>>>(cb, wsBF);
    rq_prep_norm<<<(NCB * K_CODES) / THREADS, THREADS, 0, stream>>>(cb, wsNm);
    rq_main_kernel<<<N_ROWS / ROWS_PER_BLK, THREADS, 0, stream>>>(z, cb, wsBF, wsNm, out);
}
